// MetaController_22445499088924
// MI455X (gfx1250) — compile-verified, matrix operand order checked by reading, not run on hardware
//
#include <hip/hip_runtime.h>

typedef __bf16 bf16;
typedef __attribute__((ext_vector_type(16))) __bf16 v16bf;
typedef __attribute__((ext_vector_type(8)))  float  v8f;
typedef __attribute__((ext_vector_type(4)))  unsigned int v4u;
typedef __attribute__((ext_vector_type(8)))  int v8i_t;
typedef __attribute__((ext_vector_type(4)))  int v4i_t;

constexpr int kB = 2;
constexpr int kN = 1024;
constexpr int kD = 1024;
constexpr int kR = 16;
constexpr int kH = 2048;
constexpr int kM   = kB * kN;      // 2048 flattened rows
constexpr int k3D  = 3 * kD;       // 3072
constexpr int kDR  = kD * kR;      // 16384

#if defined(__AMDGCN__) && __has_builtin(__builtin_amdgcn_tensor_load_to_lds)
#define HAS_TDM 1
#else
#define HAS_TDM 0
#endif

__device__ __forceinline__ float sigmoidf_(float x) { return 1.0f / (1.0f + __expf(-x)); }
__device__ __forceinline__ float gelu_(float x) {
    float x3 = x * x * x;
    return 0.5f * x * (1.0f + tanhf(0.7978845608028654f * (x + 0.044715f * x3)));
}

#define WMMA_BF16(a, b, c) \
    __builtin_amdgcn_wmma_f32_16x16x32_bf16(false, (a), false, (b), (short)0, (c), false, false)

// Low 32 bits of a generic pointer to LDS == LDS byte address (ISA 10.2 aperture map).
__device__ __forceinline__ unsigned lds_addr32(const void* p) {
    return (unsigned)(unsigned long long)(uintptr_t)p;
}

#if HAS_TDM
// 2D tile load via Tensor Data Mover: tile_w x tile_h elements of 2 bytes,
// row stride = stride_elems. Generous tensor dims (no OOB clipping).
__device__ __forceinline__ void tdm_load_2d(const void* gp, unsigned ldsaddr,
                                            unsigned tile_w, unsigned tile_h,
                                            unsigned stride_elems) {
    unsigned long long ga = (unsigned long long)(uintptr_t)gp;
    const unsigned tw = 1u << 20;
    v4u g0 = { 1u,                                  // count=1, user descriptor
               ldsaddr,                             // lds_addr
               (unsigned)ga,                        // global_addr[31:0]
               (unsigned)((ga >> 32) & 0x01FFFFFFu) | (2u << 30) }; // addr[56:32] | type=2
    v8i_t g1 = {
        (int)(1u << 16),                                        // data_size=1 (2B), mask=0
        (int)((tw & 0xFFFFu) << 16),                            // abar=0 | tensor_dim0 lo16
        (int)(((tw >> 16) & 0xFFFFu) | ((tw & 0xFFFFu) << 16)), // dim0 hi16 | dim1 lo16
        (int)(((tw >> 16) & 0xFFFFu) | ((tile_w & 0xFFFFu) << 16)), // dim1 hi16 | tile_dim0
        (int)(tile_h & 0xFFFFu),                                // tile_dim1 | tile_dim2=0
        (int)stride_elems,                                      // dim0_stride[31:0]
        0,                                                      // dim0_stride hi | dim1_stride lo
        0
    };
    v4i_t z = {0, 0, 0, 0};
#if __clang_major__ >= 23
    v8i_t z8 = {0, 0, 0, 0, 0, 0, 0, 0};
    __builtin_amdgcn_tensor_load_to_lds(g0, g1, z, z, z8, 0);
#else
    __builtin_amdgcn_tensor_load_to_lds(g0, g1, z, z, 0);
#endif
}
#endif

__device__ __forceinline__ void tdm_wait() {
#if HAS_TDM
#if __has_builtin(__builtin_amdgcn_s_wait_tensorcnt)
    __builtin_amdgcn_s_wait_tensorcnt(0);
#else
    asm volatile("s_wait_tensorcnt 0x0" ::: "memory");
#endif
#endif
}

// ---------------------------------------------------------------- conversions
__global__ void k_f32_to_bf16(const float* __restrict__ src, bf16* __restrict__ dst, long n) {
    long i = (long)blockIdx.x * blockDim.x + threadIdx.x;
    if (i < n) dst[i] = (bf16)src[i];
}

// W2s[r,h] = sum_d dec_w2[DR + d*R + r, h];  b2s[r] = sum_d dec_b2[DR + d*R + r]
__global__ void k_w2_reduce(const float* __restrict__ w2, const float* __restrict__ b2,
                            float* __restrict__ W2s, float* __restrict__ b2s) {
    int tid = blockIdx.x * blockDim.x + threadIdx.x;   // 16*2048
    int r = tid >> 11, h = tid & (kH - 1);
    float acc = 0.f;
    for (int d = 0; d < kD; ++d)
        acc += w2[(size_t)(kDR + d * kR + r) * kH + h];
    W2s[r * kH + h] = acc;
    if (h == 0) {
        float bb = 0.f;
        for (int d = 0; d < kD; ++d) bb += b2[kDR + d * kR + r];
        b2s[r] = bb;
    }
}

// ---------------------------------------------------------------- generic WMMA GEMM
// C[M x Nc] = A(bf16, MxK rowmajor) @ B(bf16, NcxK rowmajor)^T + bias[col]
// Block tile 128x64, BK=64, 8 waves in 4(M)x2(N), wave tile 32x32 (4 accumulators).
// EPI 0: store f32.   EPI 1: gelu, store bf16.
template <int EPI>
__global__ __launch_bounds__(256) void k_gemm(const bf16* __restrict__ A,
                                              const bf16* __restrict__ Bm,
                                              const float* __restrict__ bias,
                                              void* __restrict__ Cout,
                                              int M, int Nc, int K) {
    __shared__ __align__(32) bf16 As[2][128 * 64];   // 16KB x2
    __shared__ __align__(32) bf16 Bs[2][64 * 64];    //  8KB x2
    const int bm = blockIdx.y * 128;
    const int bn = blockIdx.x * 64;
    const int tid = threadIdx.x, lane = tid & 31, wave = tid >> 5;
    const int wm = (wave & 3) * 32;        // 4 M-waves
    const int wn = (wave >> 2) * 32;       // 2 N-waves
    v8f acc00 = {}, acc01 = {}, acc10 = {}, acc11 = {};
    const int fr = lane & 15, fkb = (lane >> 4) * 16;
#if HAS_TDM
    if (wave == 0) {
        tdm_load_2d(A  + (size_t)bm * K, lds_addr32(&As[0][0]), 64, 128, (unsigned)K);
        tdm_load_2d(Bm + (size_t)bn * K, lds_addr32(&Bs[0][0]), 64, 64,  (unsigned)K);
        tdm_wait();
    }
    __syncthreads();
    for (int kk = 0; kk < K; kk += 64) {
        const int cur = (kk >> 6) & 1, nxt = cur ^ 1;
        if (wave == 0 && kk + 64 < K) {    // async prefetch of next K-slab
            tdm_load_2d(A  + (size_t)bm * K + kk + 64, lds_addr32(&As[nxt][0]), 64, 128, (unsigned)K);
            tdm_load_2d(Bm + (size_t)bn * K + kk + 64, lds_addr32(&Bs[nxt][0]), 64, 64,  (unsigned)K);
        }
        #pragma unroll
        for (int ks = 0; ks < 64; ks += 32) {
            v16bf a0 = *(const v16bf*)&As[cur][(wm + fr)      * 64 + ks + fkb];
            v16bf a1 = *(const v16bf*)&As[cur][(wm + 16 + fr) * 64 + ks + fkb];
            v16bf b0 = *(const v16bf*)&Bs[cur][(wn + fr)      * 64 + ks + fkb];
            v16bf b1 = *(const v16bf*)&Bs[cur][(wn + 16 + fr) * 64 + ks + fkb];
            acc00 = WMMA_BF16(a0, b0, acc00);
            acc01 = WMMA_BF16(a0, b1, acc01);
            acc10 = WMMA_BF16(a1, b0, acc10);
            acc11 = WMMA_BF16(a1, b1, acc11);
        }
        if (wave == 0) tdm_wait();
        __syncthreads();
    }
#else
    for (int kk = 0; kk < K; kk += 64) {
        {   // A: 128x64 = 8192 elems, 32 per thread; B: 64x64 = 4096 elems, 16 per thread
            int ia = tid * 32, ra = ia >> 6, ca = ia & 63;
            const bf16* sa = A + (size_t)(bm + ra) * K + kk + ca;
            *(uint4*)&As[0][ra * 64 + ca]      = *(const uint4*)(sa);
            *(uint4*)&As[0][ra * 64 + ca + 8]  = *(const uint4*)(sa + 8);
            *(uint4*)&As[0][ra * 64 + ca + 16] = *(const uint4*)(sa + 16);
            *(uint4*)&As[0][ra * 64 + ca + 24] = *(const uint4*)(sa + 24);
            int ib = tid * 16, rb = ib >> 6, cb = ib & 63;
            const bf16* sb = Bm + (size_t)(bn + rb) * K + kk + cb;
            *(uint4*)&Bs[0][rb * 64 + cb]     = *(const uint4*)(sb);
            *(uint4*)&Bs[0][rb * 64 + cb + 8] = *(const uint4*)(sb + 8);
        }
        __syncthreads();
        #pragma unroll
        for (int ks = 0; ks < 64; ks += 32) {
            v16bf a0 = *(const v16bf*)&As[0][(wm + fr)      * 64 + ks + fkb];
            v16bf a1 = *(const v16bf*)&As[0][(wm + 16 + fr) * 64 + ks + fkb];
            v16bf b0 = *(const v16bf*)&Bs[0][(wn + fr)      * 64 + ks + fkb];
            v16bf b1 = *(const v16bf*)&Bs[0][(wn + 16 + fr) * 64 + ks + fkb];
            acc00 = WMMA_BF16(a0, b0, acc00);
            acc01 = WMMA_BF16(a0, b1, acc01);
            acc10 = WMMA_BF16(a1, b0, acc10);
            acc11 = WMMA_BF16(a1, b1, acc11);
        }
        __syncthreads();
    }
#endif
    const int hi8 = (lane >> 4) * 8, colL = lane & 15;
    const v8f* accs[2][2] = { { &acc00, &acc01 }, { &acc10, &acc11 } };
    for (int mi = 0; mi < 2; ++mi)
        for (int ni = 0; ni < 2; ++ni)
            for (int j = 0; j < 8; ++j) {
                int row = bm + wm + mi * 16 + j + hi8;
                int c = bn + wn + ni * 16 + colL;
                float v = (*accs[mi][ni])[j] + bias[c];
                if (EPI == 0) ((float*)Cout)[(size_t)row * Nc + c] = v;
                else          ((bf16*)Cout)[(size_t)row * Nc + c] = (bf16)gelu_(v);
            }
}

// ---------------------------------------------------------------- persistent GRU
#define GRU_BLOCKS 24
__device__ __forceinline__ void gbar(int* cnt, int* gen) {
    __syncthreads();
    if (threadIdx.x == 0) {
        __threadfence();
        int g = ((volatile int*)gen)[0];
        if (atomicAdd(cnt, 1) == GRU_BLOCKS - 1) {
            ((volatile int*)cnt)[0] = 0;
            __threadfence();
            atomicAdd(gen, 1);
        } else {
            while (((volatile int*)gen)[0] == g) __builtin_amdgcn_s_sleep(2);
        }
        __threadfence();
    }
    __syncthreads();
}

__global__ __launch_bounds__(256) void k_gru(const float* __restrict__ xp,
                                             const float* __restrict__ whh,
                                             const float* __restrict__ bhh,
                                             const float* __restrict__ beta_w,
                                             float* __restrict__ h, float* __restrict__ hp,
                                             float* __restrict__ beta, float* __restrict__ partial,
                                             int* cnt, int* gen) {
    const int tg = blockIdx.x * 256 + threadIdx.x;     // 0..6143
    __shared__ float red[256];
    if (tg < kB * kD) h[tg] = 0.f;
    gbar(cnt, gen);
    for (int t = 0; t < kN; ++t) {
        {   // hp[b, j] = h[b,:] . whh[j,:] + bhh[j]
            const int b = tg / k3D, j = tg - b * k3D;
            const float* hb = h + b * kD;
            const float* w  = whh + (size_t)j * kD;
            float s = bhh[j];
            for (int k = 0; k < kD; k += 4)
                s += hb[k]*w[k] + hb[k+1]*w[k+1] + hb[k+2]*w[k+2] + hb[k+3]*w[k+3];
            hp[b * k3D + j] = s;
        }
        gbar(cnt, gen);
        float prod = 0.f;
        if (tg < kB * kD) {
            const int b = tg >> 10, d = tg & (kD - 1);
            const float* x  = xp + ((size_t)b * kN + t) * k3D;
            const float* hh = hp + b * k3D;
            float r = sigmoidf_(x[d]        + hh[d]);
            float z = sigmoidf_(x[kD + d]   + hh[kD + d]);
            float n = tanhf   (x[2*kD + d]  + r * hh[2*kD + d]);
            float hnew = (1.f - z) * n + z * h[tg];
            h[tg] = hnew;
            prod = hnew * beta_w[d];
        }
        red[threadIdx.x] = prod;
        __syncthreads();
        for (int s2 = 128; s2 > 0; s2 >>= 1) {
            if (threadIdx.x < s2) red[threadIdx.x] += red[threadIdx.x + s2];
            __syncthreads();
        }
        if (threadIdx.x == 0 && blockIdx.x < 8) partial[blockIdx.x] = red[0];
        gbar(cnt, gen);
        if (tg < kB) {   // fixed-order sum -> deterministic
            float s = partial[tg*4] + partial[tg*4+1] + partial[tg*4+2] + partial[tg*4+3];
            beta[tg * kN + t] = sigmoidf_(s);
        }
    }
}

// ---------------------------------------------------------------- scalar-decay scan
__global__ void k_scan(const float* __restrict__ latent, const float* __restrict__ beta,
                       float* __restrict__ gated, bf16* __restrict__ gated_bf) {
    int tid = blockIdx.x * blockDim.x + threadIdx.x;   // 2048 = B*D chains
    int b = tid >> 10, d = tid & (kD - 1);
    const float* lp = latent + (size_t)b * kN * kD + d;
    const float* bp = beta + b * kN;
    float g = 0.f;
    for (int n = 0; n < kN; ++n) {
        g = lp[(size_t)n * kD] + bp[n] * g;
        size_t idx = ((size_t)b * kN + n) * kD + d;
        gated[idx] = g;
        gated_bf[idx] = (bf16)g;
    }
}

// ---------------------------------------------------------------- s[row,r] = hid . W2s[r] + b2s[r]
__global__ void k_s(const bf16* __restrict__ hid, const float* __restrict__ W2s,
                    const float* __restrict__ b2s, float* __restrict__ s) {
    int tid = blockIdx.x * blockDim.x + threadIdx.x;   // 2048*16
    int row = tid >> 4, r = tid & 15;
    const bf16* hpr = hid + (size_t)row * kH;
    const float* wp = W2s + (size_t)r * kH;
    float acc = b2s[r];
    for (int k = 0; k < kH; ++k) acc += (float)hpr[k] * wp[k];
    s[row * kR + r] = acc;
}

// ---------------------------------------------------------------- fused final GEMM
// out[row,d] = latent + gated * sum_r s[row,r] * (hid[row,:].w2a[d*16+r,:] + b2[d*16+r])
__global__ __launch_bounds__(256) void k_final(const bf16* __restrict__ hid,
                                               const bf16* __restrict__ w2a,
                                               const float* __restrict__ b2,
                                               const float* __restrict__ sarr,
                                               const float* __restrict__ gated,
                                               const float* __restrict__ latent,
                                               float* __restrict__ out) {
    __shared__ __align__(32) bf16 As[2][16 * 64];       // 2KB x2
    const int m0 = blockIdx.y * 16;
    const int d0 = blockIdx.x * 64;
    const int tid = threadIdx.x, lane = tid & 31, wave = tid >> 5;
    const int dw = d0 + wave * 8;                       // 8 d-tiles per wave
    const int hi8 = (lane >> 4) * 8, rcol = lane & 15;
    const int fr = lane & 15, fkb = (lane >> 4) * 16;
    v8f acc[8] = {};
    v8f sv;
    for (int j = 0; j < 8; ++j) sv[j] = sarr[(size_t)(m0 + j + hi8) * kR + rcol];
#if HAS_TDM
    const bf16* Abase = hid + (size_t)m0 * kH;
    if (wave == 0) {
        tdm_load_2d(Abase, lds_addr32(&As[0][0]), 64, 16, kH);
        tdm_wait();
    }
    __syncthreads();
#endif
    for (int kk = 0; kk < kH; kk += 64) {
        const int cur = (kk >> 6) & 1, nxt = cur ^ 1;
#if HAS_TDM
        if (wave == 0 && kk + 64 < kH)
            tdm_load_2d(Abase + kk + 64, lds_addr32(&As[nxt][0]), 64, 16, kH);
#else
        if (tid < 128) {    // 16x64 = 1024 elems, 8 per thread
            int lr = (tid * 8) >> 6, lc = (tid * 8) & 63;
            *(uint4*)&As[0][lr * 64 + lc] = *(const uint4*)(hid + (size_t)(m0 + lr) * kH + kk + lc);
        }
        __syncthreads();
#endif
        #pragma unroll
        for (int ks = 0; ks < 64; ks += 32) {
#if HAS_TDM
            v16bf a = *(const v16bf*)&As[cur][fr * 64 + ks + fkb];
#else
            v16bf a = *(const v16bf*)&As[0][fr * 64 + ks + fkb];
#endif
            #pragma unroll
            for (int dd = 0; dd < 8; ++dd) {
                const bf16* bp = w2a + (size_t)((dw + dd) * 16 + fr) * kH + kk + ks + fkb;
                if (kk + 64 < kH) __builtin_prefetch(bp + 64, 0, 1);
                v16bf b = *(const v16bf*)bp;
                acc[dd] = WMMA_BF16(a, b, acc[dd]);
            }
        }
#if HAS_TDM
        if (wave == 0) tdm_wait();
#endif
        __syncthreads();
    }
    for (int dd = 0; dd < 8; ++dd) {
        const int d = dw + dd;
        const float b2l = b2[d * kR + rcol];            // lane holds column r = rcol
        float t[8];
        for (int j = 0; j < 8; ++j) {
            float v = (acc[dd][j] + b2l) * sv[j];       // weight by s, fold bias
            v += __shfl_xor(v, 1);
            v += __shfl_xor(v, 2);
            v += __shfl_xor(v, 4);
            v += __shfl_xor(v, 8);                      // 16-lane halves reduce over r
            t[j] = v;
        }
        if ((lane & 15) == 0) {
            for (int j = 0; j < 8; ++j) {
                size_t idx = (size_t)(m0 + j + hi8) * kD + d;
                out[idx] = latent[idx] + gated[idx] * t[j];
            }
        }
    }
}

// ---------------------------------------------------------------- host
static inline size_t alignup(size_t x) { return (x + 255) & ~(size_t)255; }

extern "C" void kernel_launch(void* const* d_in, const int* in_sizes, int n_in,
                              void* d_out, int out_size, void* d_ws, size_t ws_size,
                              hipStream_t stream) {
    const float* latent  = (const float*)d_in[0];
    const float* w_ih    = (const float*)d_in[1];
    const float* w_hh    = (const float*)d_in[2];
    const float* b_ih    = (const float*)d_in[3];
    const float* b_hh    = (const float*)d_in[4];
    const float* beta_w  = (const float*)d_in[5];
    const float* dec_w1  = (const float*)d_in[6];
    const float* dec_b1  = (const float*)d_in[7];
    const float* dec_w2  = (const float*)d_in[8];
    const float* dec_b2  = (const float*)d_in[9];
    float* out = (float*)d_out;

    char* p = (char*)d_ws;
    size_t off = 0;
    auto carve = [&](size_t bytes) { char* r = p + off; off = alignup(off + bytes); return (void*)r; };
    bf16*  lat_bf   = (bf16*) carve((size_t)kM * kD * 2);
    bf16*  wih_bf   = (bf16*) carve((size_t)k3D * kD * 2);
    bf16*  w1_bf    = (bf16*) carve((size_t)kH * kD * 2);
    bf16*  w2a_bf   = (bf16*) carve((size_t)kDR * kH * 2);
    float* W2s      = (float*)carve((size_t)kR * kH * 4);
    float* b2s      = (float*)carve(kR * 4);
    float* xp       = (float*)carve((size_t)kM * k3D * 4);
    float* h        = (float*)carve((size_t)kB * kD * 4);
    float* hp       = (float*)carve((size_t)kB * k3D * 4);
    float* beta     = (float*)carve((size_t)kM * 4);
    float* partial  = (float*)carve(16 * 4);
    int*   bar      = (int*)  carve(2 * 4);
    float* gated    = (float*)carve((size_t)kM * kD * 4);
    bf16*  gated_bf = (bf16*) carve((size_t)kM * kD * 2);
    bf16*  hid_bf   = (bf16*) carve((size_t)kM * kH * 2);
    float* sbuf     = (float*)carve((size_t)kM * kR * 4);
    (void)ws_size; (void)n_in; (void)in_sizes; (void)out_size;

    hipMemsetAsync(bar, 0, 2 * sizeof(int), stream);

    auto cvt = [&](const float* s, bf16* d, long n) {
        k_f32_to_bf16<<<dim3((unsigned)((n + 255) / 256)), 256, 0, stream>>>(s, d, n);
    };
    cvt(latent, lat_bf, (long)kM * kD);
    cvt(w_ih,   wih_bf, (long)k3D * kD);
    cvt(dec_w1, w1_bf,  (long)kH * kD);
    cvt(dec_w2, w2a_bf, (long)kDR * kH);          // first DR rows = w1-half, contiguous
    k_w2_reduce<<<dim3(kR * kH / 256), 256, 0, stream>>>(dec_w2, dec_b2, W2s, b2s);

    // xp = latent @ w_ih^T + b_ih     (2048 x 3072, K=1024)
    k_gemm<0><<<dim3(k3D / 64, kM / 128), 256, 0, stream>>>(lat_bf, wih_bf, b_ih, xp, kM, k3D, kD);

    // sequential GRU + beta
    k_gru<<<dim3(GRU_BLOCKS), 256, 0, stream>>>(xp, w_hh, b_hh, beta_w, h, hp, beta, partial, bar, bar + 1);

    // scalar-decay scan
    k_scan<<<dim3(kB * kD / 256), 256, 0, stream>>>(latent, beta, gated, gated_bf);

    // hid = gelu(gated @ dec_w1^T + b1)   (2048 x 2048, K=1024), stored bf16
    k_gemm<1><<<dim3(kH / 64, kM / 128), 256, 0, stream>>>(gated_bf, w1_bf, dec_b1, hid_bf, kM, kH, kD);

    // s[row,r] from reduced w2
    k_s<<<dim3(kM * kR / 256), 256, 0, stream>>>(hid_bf, W2s, b2s, sbuf);

    // fused w1-GEMM + r-reduction + output
    k_final<<<dim3(kD / 64, kM / 16), 256, 0, stream>>>(hid_bf, w2a_bf, dec_b2, sbuf, gated, latent, out);
}
